// JointMamba_68247030333691
// MI455X (gfx1250) — compile-verified, matrix operand order checked by reading, not run on hardware
//
#include <hip/hip_runtime.h>
#include <hip/hip_bf16.h>
#include <math.h>

typedef __attribute__((ext_vector_type(16))) _Float16 v16h;
typedef __attribute__((ext_vector_type(8)))  _Float16 v8h;
typedef __attribute__((ext_vector_type(8)))  float    v8f;

#define BJ    2
#define CJ    256
#define H8J   64
#define W8J   64
#define LJ    2048          // (H8/2)*W8 sequence length per direction
#define DIJ   512
#define NSJ   16
#define DTRJ  16
#define MRJ   4096          // B*L rows per direction

// ---------------------------------------------------------------------------
// Register-blocked batched WMMA GEMM: D[M,N] = A[M,K]*B[N,K]^T (+resid)
// Each wave computes a 64x32 macro-tile: 4 M-frags x 2 N-frags -> 8 WMMAs
// per K-step; every fragment load is reused 2-4x (AI ~21 FLOP/B from L2).
// Requires M%64==0, N%32==0, K%32==0.
// ---------------------------------------------------------------------------
__global__ void __launch_bounds__(256)
jm_gemm_wmma_blk(const _Float16* __restrict__ A, const _Float16* __restrict__ Bw,
                 float* __restrict__ D, const float* __restrict__ resid,
                 int M, int N, int K,
                 long long sA, long long sB, long long sD)
{
    const int batch = blockIdx.z;
    const _Float16* Ab = A  + (long long)batch * sA;
    const _Float16* Bb = Bw + (long long)batch * sB;
    float*          Db = D  + (long long)batch * sD;
    const float*    Rb = resid ? resid + (long long)batch * sD : (const float*)0;

    const int jobsN = N >> 5;                 // 32 columns per job
    const int jobsM = M >> 6;                 // 64 rows per job
    const int job = blockIdx.x * (blockDim.x >> 5) + (threadIdx.x >> 5);
    if (job >= jobsM * jobsN) return;         // wave-uniform: EXEC stays full
    const int jm = job / jobsN;
    const int jn = job % jobsN;
    const int lane  = threadIdx.x & 31;
    const int lrow  = lane & 15;
    const int khalf = (lane >> 4) << 3;       // 0 or 8

    const _Float16* Ap0 = Ab + (size_t)(jm * 64 + lrow) * K + khalf;
    const _Float16* Bp0 = Bb + (size_t)(jn * 32 + lrow) * K + khalf;
    const size_t aStep = (size_t)16 * K;      // next 16-row fragment

    v8f acc[8] = {};                          // acc[mi*2+ni]
    for (int k0 = 0; k0 < K; k0 += 32) {
        v16h af[4], bf[2];
#pragma unroll
        for (int mi = 0; mi < 4; ++mi) {
            v8h lo = *(const v8h*)(Ap0 + (size_t)mi * aStep + k0);
            v8h hi = *(const v8h*)(Ap0 + (size_t)mi * aStep + k0 + 16);
#pragma unroll
            for (int e = 0; e < 8; ++e) { af[mi][e] = lo[e]; af[mi][8 + e] = hi[e]; }
        }
#pragma unroll
        for (int ni = 0; ni < 2; ++ni) {
            v8h lo = *(const v8h*)(Bp0 + (size_t)ni * aStep + k0);
            v8h hi = *(const v8h*)(Bp0 + (size_t)ni * aStep + k0 + 16);
#pragma unroll
            for (int e = 0; e < 8; ++e) { bf[ni][e] = lo[e]; bf[ni][8 + e] = hi[e]; }
        }
        __builtin_prefetch(Ap0 + k0 + 64, 0, 1);   // global_prefetch_b8
        __builtin_prefetch(Bp0 + k0 + 64, 0, 1);
#pragma unroll
        for (int mi = 0; mi < 4; ++mi)
#pragma unroll
            for (int ni = 0; ni < 2; ++ni)
                acc[mi * 2 + ni] = __builtin_amdgcn_wmma_f32_16x16x32_f16(
                    false, af[mi], false, bf[ni], (short)0, acc[mi * 2 + ni],
                    false, false);
    }

    const int colb = jn * 32 + lrow;
    const int rowb = jm * 64 + ((lane >> 4) << 3);
#pragma unroll
    for (int mi = 0; mi < 4; ++mi)
#pragma unroll
        for (int ni = 0; ni < 2; ++ni) {
            const int col = colb + ni * 16;
#pragma unroll
            for (int r = 0; r < 8; ++r) {
                const size_t o = (size_t)(rowb + mi * 16 + r) * N + col;
                float v = acc[mi * 2 + ni][r];
                if (Rb) v += Rb[o];
                Db[o] = v;
            }
        }
}

// ---------------------------------------------------------------------------
// Simple WMMA GEMM (one 16x16 tile per wave) for small-N cases (x_proj N=48).
// ---------------------------------------------------------------------------
__global__ void __launch_bounds__(256)
jm_gemm_wmma_f16(const _Float16* __restrict__ A, const _Float16* __restrict__ Bw,
                 float* __restrict__ D, int M, int N, int K,
                 long long sA, long long sB, long long sD)
{
    const int batch = blockIdx.z;
    const _Float16* Ab = A  + (long long)batch * sA;
    const _Float16* Bb = Bw + (long long)batch * sB;
    float*          Db = D  + (long long)batch * sD;

    const int tilesN = N >> 4;
    const int tilesM = M >> 4;
    const int wave = blockIdx.x * (blockDim.x >> 5) + (threadIdx.x >> 5);
    if (wave >= tilesM * tilesN) return;
    const int tm = wave / tilesN;
    const int tn = wave % tilesN;
    const int lane  = threadIdx.x & 31;
    const int lrow  = lane & 15;
    const int khalf = (lane >> 4) << 3;

    const _Float16* Ap = Ab + (size_t)(tm * 16 + lrow) * K + khalf;
    const _Float16* Bp = Bb + (size_t)(tn * 16 + lrow) * K + khalf;

    v8f acc = {};
    for (int k0 = 0; k0 < K; k0 += 32) {
        v8h alo = *(const v8h*)(Ap + k0);
        v8h ahi = *(const v8h*)(Ap + k0 + 16);
        v8h blo = *(const v8h*)(Bp + k0);
        v8h bhi = *(const v8h*)(Bp + k0 + 16);
        __builtin_prefetch(Ap + k0 + 64, 0, 1);
        __builtin_prefetch(Bp + k0 + 64, 0, 1);
        v16h a, b;
#pragma unroll
        for (int i = 0; i < 8; ++i) {
            a[i] = alo[i]; a[8 + i] = ahi[i];
            b[i] = blo[i]; b[8 + i] = bhi[i];
        }
        acc = __builtin_amdgcn_wmma_f32_16x16x32_f16(
                  false, a, false, b, (short)0, acc, false, false);
    }

    const int col  = tn * 16 + lrow;
    const int row0 = tm * 16 + ((lane >> 4) << 3);
#pragma unroll
    for (int r = 0; r < 8; ++r)
        Db[(size_t)(row0 + r) * N + col] = acc[r];
}

// ---------------------------------------------------------------------------
// f32 -> f16 weight conversion
// ---------------------------------------------------------------------------
__global__ void jm_convert_f16(const float* __restrict__ src,
                               _Float16* __restrict__ dst, long long n)
{
    long long i = (long long)blockIdx.x * blockDim.x + threadIdx.x;
    if (i < n) dst[i] = (_Float16)src[i];
}

// ---------------------------------------------------------------------------
// Fused scan_jego gather + LayerNorm. One block per (k,b,l), 256 threads = C.
// ---------------------------------------------------------------------------
__global__ void __launch_bounds__(256)
jm_gather_ln(const float* __restrict__ f0, const float* __restrict__ f1,
             const float* __restrict__ nw, const float* __restrict__ nb,
             float* __restrict__ xg, _Float16* __restrict__ xln)
{
    const int l = blockIdx.x, b = blockIdx.y, k = blockIdx.z;
    const int c = threadIdx.x;

    int h, w; const float* src;
    if (k == 0) {
        int r = l >> 6, cc = l & 63; h = 2 * r; int wc = 2 * cc;
        if (wc < W8J) { src = f0; w = wc; } else { src = f1; w = wc - W8J; }
    } else if (k == 1) {
        int i = l >> 6, j = l & 63; int hc = 2 * j + 1; w = 2 * i + 1;
        if (hc < H8J) { src = f0; h = hc; } else { src = f1; h = hc - H8J; }
    } else if (k == 2) {
        int lp = (LJ - 1) - l; int r = lp >> 6, cc = lp & 63;
        h = 2 * r; int wc = 2 * cc + 1;
        if (wc < W8J) { src = f0; w = wc; } else { src = f1; w = wc - W8J; }
    } else {
        int lp = (LJ - 1) - l; int i = lp >> 6, j = lp & 63;
        int hc = 2 * j + 1; w = 2 * i;
        if (hc < H8J) { src = f0; h = hc; } else { src = f1; h = hc - H8J; }
    }
    const float v = src[(((size_t)b * CJ + c) * H8J + h) * W8J + w];

    __shared__ float red[256];
    red[c] = v; __syncthreads();
    for (int s = 128; s > 0; s >>= 1) { if (c < s) red[c] += red[c + s]; __syncthreads(); }
    const float mu = red[0] * (1.0f / CJ); __syncthreads();
    const float dv = v - mu;
    red[c] = dv * dv; __syncthreads();
    for (int s = 128; s > 0; s >>= 1) { if (c < s) red[c] += red[c + s]; __syncthreads(); }
    const float var = red[0] * (1.0f / CJ);
    const float rs = rsqrtf(var + 1e-5f);

    const size_t o = ((size_t)((k * BJ + b) * LJ + l)) * CJ + c;
    xg[o]  = v;
    xln[o] = (_Float16)(dv * rs * nw[k * CJ + c] + nb[k * CJ + c]);
}

// ---------------------------------------------------------------------------
// Depthwise causal conv (width 4) + bias + SiLU on xa half of xz.
// ---------------------------------------------------------------------------
__global__ void jm_dwconv_silu(const float* __restrict__ D1,
                               const float* __restrict__ cw,
                               const float* __restrict__ cb,
                               float* __restrict__ u32v, _Float16* __restrict__ u16v)
{
    long long id = (long long)blockIdx.x * blockDim.x + threadIdx.x;
    if (id >= (long long)4 * MRJ * DIJ) return;
    const int d = (int)(id % DIJ);
    const int m = (int)((id / DIJ) % MRJ);
    const int k = (int)(id / ((long long)DIJ * MRJ));
    const int l = m & (LJ - 1);
    const int b = m >> 11;

    const float* wv = cw + ((size_t)k * DIJ + d) * 4;
    float acc = cb[k * DIJ + d];
#pragma unroll
    for (int j = 0; j < 4; ++j) {
        const int ll = l - 3 + j;
        if (ll >= 0)
            acc += wv[j] * D1[((size_t)k * MRJ + (b << 11) + ll) * 1024 + d];
    }
    const float s = acc * (1.0f / (1.0f + __expf(-acc)));
    const size_t o = ((size_t)k * MRJ + m) * DIJ + d;
    u32v[o] = s;
    u16v[o] = (_Float16)s;
}

// ---------------------------------------------------------------------------
// dt projection (K=16 dot) + softplus
// ---------------------------------------------------------------------------
__global__ void jm_dtproj(const float* __restrict__ D2, const float* __restrict__ dtw,
                          const float* __restrict__ dtb, float* __restrict__ dtf)
{
    long long id = (long long)blockIdx.x * blockDim.x + threadIdx.x;
    if (id >= (long long)4 * MRJ * DIJ) return;
    const int d = (int)(id % DIJ);
    const int m = (int)((id / DIJ) % MRJ);
    const int k = (int)(id / ((long long)DIJ * MRJ));

    const float* row = D2 + ((size_t)k * MRJ + m) * 48;
    const float* wv  = dtw + ((size_t)k * DIJ + d) * DTRJ;
    float acc = dtb[k * DIJ + d];
#pragma unroll
    for (int i = 0; i < DTRJ; ++i) acc += row[i] * wv[i];
    const float sp = (acc > 20.0f) ? acc : log1pf(__expf(acc));
    dtf[((size_t)k * MRJ + m) * DIJ + d] = sp;
}

// ---------------------------------------------------------------------------
// Selective scan: 16 lanes = one (k,b,d) channel's N states, 2 channels/wave.
// Fused with D*u skip and SiLU(z) gating; writes f16 for out_proj.
// ---------------------------------------------------------------------------
__global__ void __launch_bounds__(256)
jm_scan(const float* __restrict__ dtf, const float* __restrict__ u32v,
        const float* __restrict__ D2, const float* __restrict__ D1,
        const float* __restrict__ A_log, const float* __restrict__ Dpv,
        _Float16* __restrict__ yg16)
{
    const int gt   = blockIdx.x * blockDim.x + threadIdx.x;
    const int wave = gt >> 5;
    const int lane = threadIdx.x & 31;
    const int n    = lane & 15;
    const int sub  = lane >> 4;
    const int ch   = wave * 2 + sub;          // 0 .. 4095
    const int d    = ch & (DIJ - 1);
    const int kb   = ch >> 9;
    const int b    = kb & (BJ - 1);
    const int k    = kb >> 1;

    const float Ad  = -__expf(A_log[((size_t)k * DIJ + d) * NSJ + n]);
    const float Dpc = Dpv[k * DIJ + d];
    float h = 0.0f;

    for (int t = 0; t < LJ; ++t) {
        const size_t mb = (size_t)k * MRJ + (b << 11) + t;
        const float dt = dtf[mb * DIJ + d];
        const float ut = u32v[mb * DIJ + d];
        const float Bn = D2[mb * 48 + 16 + n];
        const float Cn = D2[mb * 48 + 32 + n];
        h = __expf(dt * Ad) * h + dt * Bn * ut;
        float p = h * Cn;
        p += __shfl_xor(p, 1);
        p += __shfl_xor(p, 2);
        p += __shfl_xor(p, 4);
        p += __shfl_xor(p, 8);
        if (n == 0) {
            const float y = p + Dpc * ut;
            const float z = D1[mb * 1024 + DIJ + d];
            const float g = y * z * (1.0f / (1.0f + __expf(-z)));
            yg16[mb * DIJ + d] = (_Float16)g;
        }
    }
}

// ---------------------------------------------------------------------------
// merge_jego as pure permutation gather -> conv input X [4, C, 64, 64]
// ---------------------------------------------------------------------------
__global__ void jm_merge(const float* __restrict__ yf, float* __restrict__ X)
{
    long long id = (long long)blockIdx.x * blockDim.x + threadIdx.x;
    if (id >= (long long)4 * CJ * 4096) return;
    const int hw   = (int)(id & 4095);
    const int c    = (int)((id >> 12) & (CJ - 1));
    const int nidx = (int)(id >> 20);
    const int h = hw >> 6, w = hw & 63;
    const int dsel = nidx >> 1;
    const int b    = nidx & 1;

    int k, l;
    if ((h & 1) == 0) {                       // y2w branch
        const int r = h >> 1;
        if ((w & 1) == 0) { const int c2 = (w >> 1) + (dsel ? 32 : 0);
                            k = 0; l = r * 64 + c2; }
        else              { const int c2 = ((w - 1) >> 1) + (dsel ? 32 : 0);
                            k = 2; l = (LJ - 1) - (r * 64 + c2); }
    } else {                                  // y2h branch
        const int j = ((h - 1) >> 1) + (dsel ? 32 : 0);
        if (w & 1) { const int i = (w - 1) >> 1; k = 1; l = i * 64 + j; }
        else       { const int i = w >> 1;       k = 3; l = (LJ - 1) - (i * 64 + j); }
    }
    X[id] = yf[((size_t)((k * BJ + b) * LJ + l)) * CJ + c];
}

// ---------------------------------------------------------------------------
// im2col for 3x3 SAME conv: A5[row=(n,h,w)][kidx=c*9+kh*3+kw] (f16)
// ---------------------------------------------------------------------------
__global__ void jm_im2col(const float* __restrict__ X, _Float16* __restrict__ A5)
{
    long long id = (long long)blockIdx.x * blockDim.x + threadIdx.x;
    if (id >= (long long)16384 * 2304) return;
    const int kidx = (int)(id % 2304);
    const int row  = (int)(id / 2304);
    const int c  = kidx / 9;
    const int t  = kidx % 9;
    const int kh = t / 3, kw = t % 3;
    const int n  = row >> 12;
    const int hw = row & 4095;
    const int h  = hw >> 6, w = hw & 63;
    const int hh = h + kh - 1, ww = w + kw - 1;
    float v = 0.0f;
    if (hh >= 0 && hh < 64 && ww >= 0 && ww < 64)
        v = X[(((size_t)n * CJ + c) * 64 + hh) * 64 + ww];
    A5[(size_t)row * 2304 + kidx] = (_Float16)v;
}

// ---------------------------------------------------------------------------
// GLU epilogue + final layout [2, B, C, 4096]
// ---------------------------------------------------------------------------
__global__ void jm_glu_out(const float* __restrict__ D5, const float* __restrict__ gb,
                           float* __restrict__ out)
{
    long long id = (long long)blockIdx.x * blockDim.x + threadIdx.x;
    if (id >= (long long)2 * BJ * CJ * 4096) return;
    const int hw = (int)(id & 4095);
    const int c  = (int)((id >> 12) & (CJ - 1));
    const int b  = (int)((id >> 20) & (BJ - 1));
    const int s  = (int)(id >> 21);
    const int n  = s * BJ + b;
    const size_t row = (size_t)n * 4096 + hw;
    const float a = D5[row * 512 + c]       + gb[c];
    const float g = D5[row * 512 + 256 + c] + gb[256 + c];
    out[id] = a * (1.0f / (1.0f + __expf(-g)));
}

// ===========================================================================
extern "C" void kernel_launch(void* const* d_in, const int* in_sizes, int n_in,
                              void* d_out, int out_size, void* d_ws, size_t ws_size,
                              hipStream_t stream)
{
    const float* feat0  = (const float*)d_in[0];
    const float* feat1  = (const float*)d_in[1];
    const float* norm_w = (const float*)d_in[2];
    const float* norm_b = (const float*)d_in[3];
    const float* in_w   = (const float*)d_in[4];
    const float* conv_w = (const float*)d_in[5];
    const float* conv_b = (const float*)d_in[6];
    const float* xprj_w = (const float*)d_in[7];
    const float* dt_w   = (const float*)d_in[8];
    const float* dt_b   = (const float*)d_in[9];
    const float* A_log  = (const float*)d_in[10];
    const float* Dp     = (const float*)d_in[11];
    const float* out_w  = (const float*)d_in[12];
    const float* glu_w  = (const float*)d_in[13];
    const float* glu_b  = (const float*)d_in[14];
    float* out = (float*)d_out;

    // ---- workspace carving (256B aligned) ----
    char* base = (char*)d_ws;
    size_t off = 0;
    auto alloc = [&](size_t bytes) {
        void* p = base + off;
        off += (bytes + 255) & ~(size_t)255;
        return p;
    };
    _Float16* w16_in  = (_Float16*)alloc((size_t)4 * 1024 * 256 * 2);
    _Float16* w16_xp  = (_Float16*)alloc((size_t)4 * 48 * 512 * 2);
    _Float16* w16_out = (_Float16*)alloc((size_t)4 * 256 * 512 * 2);
    _Float16* w16_glu = (_Float16*)alloc((size_t)512 * 2304 * 2);
    float*    xg      = (float*)   alloc((size_t)4 * MRJ * CJ * 4);
    _Float16* xln16   = (_Float16*)alloc((size_t)4 * MRJ * CJ * 2);
    float*    D1      = (float*)   alloc((size_t)4 * MRJ * 1024 * 4);
    float*    u32v    = (float*)   alloc((size_t)4 * MRJ * DIJ * 4);
    _Float16* u16v    = (_Float16*)alloc((size_t)4 * MRJ * DIJ * 2);
    float*    D2      = (float*)   alloc((size_t)4 * MRJ * 48 * 4);
    float*    dtf     = (float*)   alloc((size_t)4 * MRJ * DIJ * 4);
    _Float16* yg16    = (_Float16*)alloc((size_t)4 * MRJ * DIJ * 2);
    float*    yf      = (float*)   alloc((size_t)4 * MRJ * CJ * 4);   // resid-fused out_proj
    float*    X       = (float*)   alloc((size_t)4 * CJ * 4096 * 4);
    _Float16* A5      = (_Float16*)alloc((size_t)16384 * 2304 * 2);
    float*    D5      = (float*)   alloc((size_t)16384 * 512 * 4);
    (void)ws_size; (void)n_in; (void)in_sizes; (void)out_size;

    auto cvt = [&](const float* s, _Float16* d, long long n) {
        jm_convert_f16<<<dim3((unsigned)((n + 255) / 256)), 256, 0, stream>>>(s, d, n);
    };
    cvt(in_w,   w16_in,  (long long)4 * 1024 * 256);
    cvt(xprj_w, w16_xp,  (long long)4 * 48 * 512);
    cvt(out_w,  w16_out, (long long)4 * 256 * 512);
    cvt(glu_w,  w16_glu, (long long)512 * 2304);

    // 1) gather + layernorm
    jm_gather_ln<<<dim3(LJ, BJ, 4), 256, 0, stream>>>(feat0, feat1, norm_w, norm_b,
                                                      xg, xln16);

    // 2) in_proj: [4] x (4096x1024x256), 64x32 macro-tiles
    {
        int jobs = (MRJ / 64) * (1024 / 32);
        jm_gemm_wmma_blk<<<dim3((jobs + 7) / 8, 1, 4), 256, 0, stream>>>(
            xln16, w16_in, D1, (const float*)0, MRJ, 1024, 256,
            (long long)MRJ * 256, (long long)1024 * 256, (long long)MRJ * 1024);
    }

    // 3) depthwise conv + SiLU
    {
        long long n = (long long)4 * MRJ * DIJ;
        jm_dwconv_silu<<<dim3((unsigned)((n + 255) / 256)), 256, 0, stream>>>(
            D1, conv_w, conv_b, u32v, u16v);
    }

    // 4) x_proj: [4] x (4096x48x512) — N=48, simple 16x16 kernel
    {
        int tiles = (MRJ / 16) * (48 / 16);
        jm_gemm_wmma_f16<<<dim3((tiles + 7) / 8, 1, 4), 256, 0, stream>>>(
            u16v, w16_xp, D2, MRJ, 48, 512,
            (long long)MRJ * 512, (long long)48 * 512, (long long)MRJ * 48);
    }

    // 5) dt projection + softplus
    {
        long long n = (long long)4 * MRJ * DIJ;
        jm_dtproj<<<dim3((unsigned)((n + 255) / 256)), 256, 0, stream>>>(
            D2, dt_w, dt_b, dtf);
    }

    // 6) selective scan + gating (4096 channels, 2 per wave => 65536 threads)
    jm_scan<<<dim3(256), 256, 0, stream>>>(dtf, u32v, D2, D1, A_log, Dp, yg16);

    // 7) out_proj + fused residual: [4] x (4096x256x512) -> yf
    {
        int jobs = (MRJ / 64) * (CJ / 32);
        jm_gemm_wmma_blk<<<dim3((jobs + 7) / 8, 1, 4), 256, 0, stream>>>(
            yg16, w16_out, yf, xg, MRJ, CJ, 512,
            (long long)MRJ * 512, (long long)CJ * 512, (long long)MRJ * CJ);
    }

    // 8) merge_jego gather -> conv input X
    {
        long long n = (long long)4 * CJ * 4096;
        jm_merge<<<dim3((unsigned)((n + 255) / 256)), 256, 0, stream>>>(yf, X);
    }

    // 9) im2col (f16)
    {
        long long n = (long long)16384 * 2304;
        jm_im2col<<<dim3((unsigned)((n + 255) / 256)), 256, 0, stream>>>(X, A5);
    }

    // 10) 3x3 GLU conv as GEMM: 16384 x 512 x 2304, 64x32 macro-tiles
    {
        int jobs = (16384 / 64) * (512 / 32);
        jm_gemm_wmma_blk<<<dim3((jobs + 7) / 8, 1, 1), 256, 0, stream>>>(
            A5, w16_glu, D5, (const float*)0, 16384, 512, 2304, 0, 0, 0);
    }

    // 11) GLU epilogue -> output [2, B, C, 4096]
    {
        long long n = (long long)2 * BJ * CJ * 4096;
        jm_glu_out<<<dim3((unsigned)((n + 255) / 256)), 256, 0, stream>>>(D5, glu_b, out);
    }
}
